// SelfAttention_71837622993180
// MI455X (gfx1250) — compile-verified, matrix operand order checked by reading, not run on hardware
//
#include <hip/hip_runtime.h>
#include <hip/hip_bf16.h>
#include <math.h>

typedef __attribute__((ext_vector_type(16))) _Float16 v16h;
typedef __attribute__((ext_vector_type(8)))  _Float16 v8h;
typedef __attribute__((ext_vector_type(8)))  float    v8f;
typedef __attribute__((ext_vector_type(4)))  unsigned int v4u;
typedef __attribute__((ext_vector_type(8)))  int      v8i;
typedef __attribute__((ext_vector_type(4)))  int      v4i;

#define B_   2
#define T_   2048
#define D_   2048
#define NH_  32
#define NKV_ 8
#define HD_  64
#define REP_ (NH_ / NKV_)
#define DKV_ (NKV_ * HD_)   // 512
#define MT_  (B_ * T_)      // 4096

static __device__ __forceinline__ v16h cat8(v8h lo, v8h hi) {
  return __builtin_shufflevector(lo, hi, 0,1,2,3,4,5,6,7,8,9,10,11,12,13,14,15);
}
static __device__ __forceinline__ v8f wmma16(v16h a, v16h b, v8f c) {
  return __builtin_amdgcn_wmma_f32_16x16x32_f16(false, a, false, b, (short)0, c, false, false);
}

// ---------------- elementwise convert f32 -> f16 ----------------
__global__ void cvt_f32_f16(const float* __restrict__ in, _Float16* __restrict__ out, size_t n) {
  size_t i = (size_t)blockIdx.x * blockDim.x + threadIdx.x;
  if (i < n) out[i] = (_Float16)in[i];
}

// ------------- transpose+convert: in (K x N) f32 -> out (N x K) f16 -------------
__global__ void trans_cvt(const float* __restrict__ in, _Float16* __restrict__ out, int K, int N) {
  size_t i = (size_t)blockIdx.x * blockDim.x + threadIdx.x;
  if (i < (size_t)K * N) {
    int n = (int)(i % (size_t)N);
    int k = (int)(i / (size_t)N);
    out[(size_t)n * K + k] = (_Float16)in[i];
  }
}

// ---------------- WMMA GEMM: C(MxN) = A(MxK,f16) @ Bt(NxK,f16)^T ----------------
// Block = 256 threads = 8 waves; wave owns 64(M) x 64(N); block tile 128 x 256.
// grid = (N/256, M/128).
// OUTMODE: 0 = f32 row-major, 1 = f16 row-major, 2 = f16 transposed (C^T, N x M).
template <int OUTMODE>
__global__ __launch_bounds__(256) void gemm_wmma(const _Float16* __restrict__ A,
                                                 const _Float16* __restrict__ Bt,
                                                 void* __restrict__ Cout,
                                                 int M, int N, int K) {
  const int lane = threadIdx.x & 31;
  const int wave = threadIdx.x >> 5;
  const int r16  = lane & 15;
  const int kh   = lane >> 4;
  const int mbase = blockIdx.y * 128 + (wave >> 2) * 64;
  const int nbase = blockIdx.x * 256 + (wave & 3) * 64;

  v8f acc[4][4] = {};
  const _Float16* arow[4];
#pragma unroll
  for (int mi = 0; mi < 4; ++mi) arow[mi] = A + (size_t)(mbase + mi * 16 + r16) * K;
  const _Float16* brow[4];
#pragma unroll
  for (int ni = 0; ni < 4; ++ni) brow[ni] = Bt + (size_t)(nbase + ni * 16 + r16) * K;

  for (int kb = 0; kb < K; kb += 32) {
    v16h bf[4];
#pragma unroll
    for (int ni = 0; ni < 4; ++ni) {
      v8h b0 = *(const v8h*)(brow[ni] + kb + 8 * kh);
      v8h b1 = *(const v8h*)(brow[ni] + kb + 16 + 8 * kh);
      bf[ni] = cat8(b0, b1);
    }
#pragma unroll
    for (int mi = 0; mi < 4; ++mi) {
      v8h a0 = *(const v8h*)(arow[mi] + kb + 8 * kh);
      v8h a1 = *(const v8h*)(arow[mi] + kb + 16 + 8 * kh);
      v16h af = cat8(a0, a1);
#pragma unroll
      for (int ni = 0; ni < 4; ++ni)
        acc[mi][ni] = wmma16(af, bf[ni], acc[mi][ni]);
    }
  }

#pragma unroll
  for (int mi = 0; mi < 4; ++mi) {
#pragma unroll
    for (int ni = 0; ni < 4; ++ni) {
      const int col = nbase + ni * 16 + r16;
      if constexpr (OUTMODE == 2) {
        // C^T: 8 consecutive rows per lane -> one 16B store
        v8h st;
#pragma unroll
        for (int v = 0; v < 8; ++v) st[v] = (_Float16)acc[mi][ni][v];
        *(v8h*)((_Float16*)Cout + (size_t)col * M + mbase + mi * 16 + 8 * kh) = st;
      } else {
#pragma unroll
        for (int v = 0; v < 8; ++v) {
          const int row = mbase + mi * 16 + 8 * kh + v;
          if constexpr (OUTMODE == 1)
            ((_Float16*)Cout)[(size_t)row * N + col] = (_Float16)acc[mi][ni][v];
          else
            ((float*)Cout)[(size_t)row * N + col] = acc[mi][ni][v];
        }
      }
    }
  }
}

// ---------------- Flash attention (causal, GQA) ----------------
// One wave per (b, h, 16-query tile). grid = (T/16, NH, B), block = 32.
// K blocks staged into LDS by the Tensor Data Mover; V read from the transposed
// Vt (d-major) layout so every fragment load is a contiguous 16B vector.
__global__ __launch_bounds__(32) void attn_fa(const _Float16* __restrict__ Q,
                                              const _Float16* __restrict__ Kc,
                                              const _Float16* __restrict__ Vt,
                                              _Float16* __restrict__ Y) {
  __shared__ __align__(16) _Float16 lds_k[32 * 64];  // 4KB K-block stage
  __shared__ __align__(16) _Float16 lds_p[16 * 32];  // 1KB P transpose bounce
  const int lane = threadIdx.x & 31;
  const int r16  = lane & 15;
  const int kh   = lane >> 4;
  const int qb   = blockIdx.x * 16;
  const int h    = blockIdx.y;
  const int b    = blockIdx.z;
  const int kvh  = h / REP_;

  // Resident Q fragments: 16 rows x 64 d
  const _Float16* qrow = Q + (size_t)(b * T_ + qb + r16) * D_ + h * HD_;
  v16h qf[2];
#pragma unroll
  for (int ks = 0; ks < 2; ++ks) {
    v8h q0 = *(const v8h*)(qrow + ks * 32 + 8 * kh);
    v8h q1 = *(const v8h*)(qrow + ks * 32 + 16 + 8 * kh);
    qf[ks] = cat8(q0, q1);
  }

  v8f o[4] = {};
  float m[8], l[8];
#pragma unroll
  for (int v = 0; v < 8; ++v) { m[v] = -3.0e38f; l[v] = 0.0f; }

  // TDM descriptor invariants (2D tensor: this head's 64-wide slice of K, T_ rows)
  const _Float16* kbase0 = Kc + (size_t)(b * T_) * DKV_ + kvh * HD_;
  const unsigned lko = (unsigned)(uintptr_t)lds_k;  // LDS byte offset (low 32 bits)
  // group1: data_size=2B; tensor_dim0=64, tensor_dim1=T_; tile 64x32; stride0=DKV_
  const v8i g1 = { (int)(1u << 16), (int)(64u << 16), (int)(((unsigned)T_) << 16),
                   (int)(64u << 16), 32, DKV_, 0, 0 };
  const v4i gz4 = { 0, 0, 0, 0 };
  const v8i gz8 = { 0, 0, 0, 0, 0, 0, 0, 0 };

  const float sc = 0.125f;  // 1/sqrt(64)
  const int jend = qb + 16;
  for (int jb = 0; jb < jend; jb += 32) {
    // make sure prior ds_loads from lds_k have completed before TDM overwrites it
    asm volatile("s_wait_dscnt 0x0" ::: "memory");
    {
      unsigned long long ga = (unsigned long long)(uintptr_t)(kbase0 + (size_t)jb * DKV_);
      v4u g0 = { 1u, lko, (unsigned)ga,
                 (unsigned)((ga >> 32) & 0x1FFFFFFull) | (2u << 30) };
      __builtin_amdgcn_tensor_load_to_lds(g0, g1, gz4, gz4, gz8, 0);
    }
    __builtin_amdgcn_s_wait_tensorcnt(0);

    // S = Q @ K^T for 32 keys (two 16-wide j tiles), K read from LDS
    v8f s0 = {}, s1 = {};
#pragma unroll
    for (int ks = 0; ks < 2; ++ks) {
      const _Float16* k0 = lds_k + r16 * 64 + ks * 32;
      const _Float16* k1 = lds_k + (16 + r16) * 64 + ks * 32;
      v16h bf0 = cat8(*(const v8h*)(k0 + 8 * kh), *(const v8h*)(k0 + 16 + 8 * kh));
      v16h bf1 = cat8(*(const v8h*)(k1 + 8 * kh), *(const v8h*)(k1 + 16 + 8 * kh));
      s0 = wmma16(qf[ks], bf0, s0);
      s1 = wmma16(qf[ks], bf1, s1);
    }

    // scale + causal mask (C layout: lane holds col j, rows v+8*kh)
    const int j0 = jb + r16, j1 = jb + 16 + r16;
#pragma unroll
    for (int v = 0; v < 8; ++v) {
      const int qp = qb + 8 * kh + v;
      s0[v] = (j0 <= qp) ? s0[v] * sc : -1.0e30f;
      s1[v] = (j1 <= qp) ? s1[v] * sc : -1.0e30f;
    }

    // online softmax: row reductions across each 16-lane group
#pragma unroll
    for (int v = 0; v < 8; ++v) {
      float t = fmaxf(s0[v], s1[v]);
#pragma unroll
      for (int off = 1; off < 16; off <<= 1) t = fmaxf(t, __shfl_xor(t, off, 16));
      const float mn = fmaxf(m[v], t);
      const float r  = __expf(m[v] - mn);
      const float p0 = __expf(s0[v] - mn);
      const float p1 = __expf(s1[v] - mn);
      float su = p0 + p1;
#pragma unroll
      for (int off = 1; off < 16; off <<= 1) su += __shfl_xor(su, off, 16);
      l[v] = l[v] * r + su;
      m[v] = mn;
#pragma unroll
      for (int dt = 0; dt < 4; ++dt) o[dt][v] *= r;
      lds_p[(8 * kh + v) * 32 + r16]      = (_Float16)p0;
      lds_p[(8 * kh + v) * 32 + 16 + r16] = (_Float16)p1;
    }

    // reload P as an A fragment (same-wave LDS ops are in-order)
    v16h pa = cat8(*(const v8h*)(&lds_p[r16 * 32 + 8 * kh]),
                   *(const v8h*)(&lds_p[r16 * 32 + 16 + 8 * kh]));

    // O += P @ V using the d-major Vt layout: contiguous 16B runs along j
#pragma unroll
    for (int dt = 0; dt < 4; ++dt) {
      const _Float16* vr = Vt + (size_t)(kvh * HD_ + dt * 16 + r16) * MT_ + b * T_ + jb;
      v16h vb = cat8(*(const v8h*)(vr + 8 * kh), *(const v8h*)(vr + 16 + 8 * kh));
      o[dt] = wmma16(pa, vb, o[dt]);
    }
  }

  // normalize and store y (f16, row = b*T+q, col = h*64+d)
#pragma unroll
  for (int v = 0; v < 8; ++v) {
    const float inv = 1.0f / l[v];
    const int row = b * T_ + qb + 8 * kh + v;
#pragma unroll
    for (int dt = 0; dt < 4; ++dt)
      Y[(size_t)row * D_ + h * HD_ + dt * 16 + r16] = (_Float16)(o[dt][v] * inv);
  }
}

// ---------------------------------------------------------------
extern "C" void kernel_launch(void* const* d_in, const int* in_sizes, int n_in,
                              void* d_out, int out_size, void* d_ws, size_t ws_size,
                              hipStream_t stream) {
  (void)in_sizes; (void)n_in; (void)out_size; (void)ws_size;
  const float* x  = (const float*)d_in[0];
  const float* wq = (const float*)d_in[1];
  const float* wk = (const float*)d_in[2];
  const float* wv = (const float*)d_in[3];
  const float* wo = (const float*)d_in[4];
  float* out = (float*)d_out;

  const size_t MT = (size_t)MT_;

  char* ws = (char*)d_ws;
  size_t off = 0;
  auto alloc = [&](size_t elems) -> _Float16* {
    _Float16* p = (_Float16*)(ws + off);
    off += elems * sizeof(_Float16);
    return p;
  };
  _Float16* x16 = alloc(MT * D_);
  _Float16* wqt = alloc((size_t)D_ * D_);
  _Float16* wkt = alloc((size_t)D_ * DKV_);
  _Float16* wvt = alloc((size_t)D_ * DKV_);
  _Float16* wot = alloc((size_t)D_ * D_);
  _Float16* Qm  = alloc(MT * D_);
  _Float16* Km  = alloc(MT * (size_t)DKV_);
  _Float16* Vtm = alloc(MT * (size_t)DKV_);  // transposed: (DKV_) x (MT)
  _Float16* Ym  = alloc(MT * D_);

  { size_t n = MT * D_;           cvt_f32_f16<<<(unsigned)((n + 255) / 256), 256, 0, stream>>>(x, x16, n); }
  { size_t n = (size_t)D_ * D_;   trans_cvt<<<(unsigned)((n + 255) / 256), 256, 0, stream>>>(wq, wqt, D_, D_); }
  { size_t n = (size_t)D_ * DKV_; trans_cvt<<<(unsigned)((n + 255) / 256), 256, 0, stream>>>(wk, wkt, D_, DKV_); }
  { size_t n = (size_t)D_ * DKV_; trans_cvt<<<(unsigned)((n + 255) / 256), 256, 0, stream>>>(wv, wvt, D_, DKV_); }
  { size_t n = (size_t)D_ * D_;   trans_cvt<<<(unsigned)((n + 255) / 256), 256, 0, stream>>>(wo, wot, D_, D_); }

  dim3 blk(256);
  gemm_wmma<1><<<dim3(D_ / 256,   (unsigned)(MT / 128)), blk, 0, stream>>>(x16, wqt, Qm,  MT_, D_,   D_);
  gemm_wmma<1><<<dim3(DKV_ / 256, (unsigned)(MT / 128)), blk, 0, stream>>>(x16, wkt, Km,  MT_, DKV_, D_);
  gemm_wmma<2><<<dim3(DKV_ / 256, (unsigned)(MT / 128)), blk, 0, stream>>>(x16, wvt, Vtm, MT_, DKV_, D_);

  attn_fa<<<dim3(T_ / 16, NH_, B_), dim3(32), 0, stream>>>(Qm, Km, Vtm, Ym);

  gemm_wmma<0><<<dim3(D_ / 256, (unsigned)(MT / 128)), blk, 0, stream>>>(Ym, wot, out, MT_, D_, D_);
}
